// DeStationaryAttention_9053791060570
// MI455X (gfx1250) — matrix operand order repaired and checked by reading, not compiled, not run on hardware
//
#include <hip/hip_runtime.h>

// ---- problem constants (reference: B=4, T=1024, N=16, D=128) ----
#define B_BATCH 4
#define T_SEQ   1024
#define N_HEADS 16
#define D_MODEL 128

#define QTILE   128     // query rows per block (8 waves x 16 rows)
#define KVTILE  32      // kv rows per inner iteration

// padded LDS strides (in bf16 elements) chosen for conflict-friendly b128 loads
#define QSTR    136     // 128 + 8  -> row stride 272B (68 dwords == 4 mod 64 banks)
#define KSTR    136
#define VTSTR   40      // Vt rows of 32 kv + pad -> 80B (20 dwords, odd-ish spread)
#define PSTR    40

typedef __bf16 bf16;
typedef __attribute__((ext_vector_type(16))) __bf16 v16bf;
typedef __attribute__((ext_vector_type(8)))  __bf16 bf16x8;   // 16 bytes
typedef __attribute__((ext_vector_type(8)))  float  v8f;

union Frag16 { v16bf v; bf16x8 h[2]; };

// ---- DPP16 butterfly reduction over each 16-lane half (VALU-only, no LDS) ----
// ctrl 0xB1 = quad_perm(1,0,3,2)  -> lane ^ 1
// ctrl 0x4E = quad_perm(2,3,0,1)  -> lane ^ 2
// ctrl 0x141 = row_half_mirror    -> lane ^ 7  (combines quads; quads already uniform)
// ctrl 0x140 = row_mirror         -> lane ^ 15 (combines 8-groups; groups already uniform)
template <int CTRL>
__device__ __forceinline__ float dpp_movf(float x) {
    return __builtin_bit_cast(float,
        __builtin_amdgcn_update_dpp(0, __builtin_bit_cast(int, x), CTRL, 0xF, 0xF, true));
}
__device__ __forceinline__ float row16_max(float v) {
    v = fmaxf(v, dpp_movf<0xB1>(v));
    v = fmaxf(v, dpp_movf<0x4E>(v));
    v = fmaxf(v, dpp_movf<0x141>(v));
    v = fmaxf(v, dpp_movf<0x140>(v));
    return v;
}
__device__ __forceinline__ float row16_sum(float v) {
    v += dpp_movf<0xB1>(v);
    v += dpp_movf<0x4E>(v);
    v += dpp_movf<0x141>(v);
    v += dpp_movf<0x140>(v);
    return v;
}

// A-matrix (16x32 bf16) fragment from row-major LDS tile.
// ISA layout: lanes 0-15 -> M=lane, K = {0..7, 16..23}; lanes 16-31 -> M=lane-16, K = {8..15, 24..31}
__device__ __forceinline__ void load_a_frag(Frag16& f, const bf16* tile, int stride,
                                            int m_lo, int kbase, int lane) {
    int m    = lane & 15;
    int koff = (lane >> 4) * 8;
    const bf16* p = tile + (m_lo + m) * stride + kbase + koff;
    f.h[0] = *(const bf16x8*)(p);        // K = koff .. koff+7
    f.h[1] = *(const bf16x8*)(p + 16);   // K = 16+koff .. 16+koff+7
}

// B-matrix (32x16 bf16) fragment where B(k,n) = src[rowbase+n][colbase+k]
// (i.e. B is the transpose of a row-major LDS tile slice).
// Layout: lane n = lane&15, lanes 0-15 hold K = {0..7, 16..23}, lanes 16-31 hold K = {8..15, 24..31}.
__device__ __forceinline__ void load_b_frag(Frag16& f, const bf16* src, int stride,
                                            int rowbase, int colbase, int lane) {
    int nn = lane & 15;
    int hi = lane >> 4;
    const bf16* p = src + (rowbase + nn) * stride + colbase + hi * 8;
    f.h[0] = *(const bf16x8*)(p);
    f.h[1] = *(const bf16x8*)(p + 16);
}

// ---------------- Kernel 1: per-head temporal means of Q,K and tau ----------------
__global__ __launch_bounds__(256)
void destat_stats_kernel(const float* __restrict__ Q, const float* __restrict__ K,
                         const float* __restrict__ stdv, const float* __restrict__ tw,
                         const float* __restrict__ tb, float* __restrict__ meanQ,
                         float* __restrict__ meanK, float* __restrict__ tau) {
    __shared__ float rq[256], rk[256], rs[256];
    const int tid  = threadIdx.x;
    const int head = blockIdx.x;          // b*N + n
    const int b    = head >> 4;
    const int n    = head & 15;
    const int d    = tid & 127;
    const int h    = tid >> 7;

    float sq = 0.f, sk = 0.f;
    for (int t = h * (T_SEQ / 2); t < (h + 1) * (T_SEQ / 2); ++t) {
        size_t g = ((size_t)(b * T_SEQ + t) * N_HEADS + n) * D_MODEL + d;
        sq += Q[g];
        sk += K[g];
    }
    rq[tid] = sq;
    rk[tid] = sk;

    float ss = 0.f;
    for (int t = tid; t < T_SEQ; t += 256)
        ss += stdv[(size_t)(b * T_SEQ + t) * N_HEADS + n];
    rs[tid] = ss;
    __syncthreads();

    if (h == 0) {
        meanQ[head * D_MODEL + d] = (rq[tid] + rq[tid + 128]) * (1.0f / T_SEQ);
        meanK[head * D_MODEL + d] = (rk[tid] + rk[tid + 128]) * (1.0f / T_SEQ);
    }
    for (int s = 128; s > 0; s >>= 1) {
        if (tid < s) rs[tid] += rs[tid + s];
        __syncthreads();
    }
    if (tid == 0) {
        float ms = rs[0] * (1.0f / T_SEQ);
        float x  = ms * tw[0] + tb[0];
        tau[head] = 2.0f / (1.0f + expf(-x));   // 2*sigmoid
    }
}

// ---------------- Kernel 2: flash attention with bf16 WMMA ----------------
__global__ __launch_bounds__(256)
void destat_attn_kernel(const float* __restrict__ Q, const float* __restrict__ K,
                        const float* __restrict__ V, const float* __restrict__ meanQ,
                        const float* __restrict__ meanK, const float* __restrict__ tau,
                        float* __restrict__ out) {
    __shared__ __attribute__((aligned(16))) bf16 Qs[QTILE * QSTR];     // 34816 B
    __shared__ __attribute__((aligned(16))) bf16 Ks[KVTILE * KSTR];    //  8704 B
    __shared__ __attribute__((aligned(16))) bf16 Vt[D_MODEL * VTSTR];  // 10240 B (V transposed)
    __shared__ __attribute__((aligned(16))) bf16 Ps[8 * 16 * PSTR];    // 10240 B (per-wave P scratch)

    const int tid  = threadIdx.x;
    const int lane = tid & 31;            // wave32
    const int wave = tid >> 5;            // 8 waves
    const int head = blockIdx.y;          // b*N + n
    const int b    = head >> 4;
    const int n    = head & 15;
    const int q0   = blockIdx.x * QTILE;

    // fold scale * tau into Q (tau > 0, softmax-safe)
    const float qscale = 0.088388347648318447f * tau[head];   // D^-0.5 * tau

    // stage centered+scaled Q tile -> LDS (coalesced global reads)
    for (int i = 0; i < (QTILE * D_MODEL) / 256; ++i) {
        int idx = i * 256 + tid;
        int r = idx >> 7, c = idx & 127;
        float v = Q[((size_t)(b * T_SEQ + q0 + r) * N_HEADS + n) * D_MODEL + c];
        Qs[r * QSTR + c] = (bf16)((v - meanQ[head * D_MODEL + c]) * qscale);
    }
    __syncthreads();

    // this wave's 16 Q rows live in registers for the whole kv loop (4 x v16bf)
    Frag16 qa[4];
#pragma unroll
    for (int kk = 0; kk < 4; ++kk)
        load_a_frag(qa[kk], Qs, QSTR, wave * 16, kk * 32, lane);

    v8f   O[8];                 // 16 rows x 128 cols of output accumulator
    float Mrow[8], Lrow[8];     // per-row running max / sum (replicated across 16-lane half)
    const v8f vzero = {0.f, 0.f, 0.f, 0.f, 0.f, 0.f, 0.f, 0.f};
#pragma unroll
    for (int c = 0; c < 8; ++c) O[c] = vzero;
#pragma unroll
    for (int j = 0; j < 8; ++j) { Mrow[j] = -1e30f; Lrow[j] = 0.0f; }

    const float LOG2E = 1.4426950408889634f;
    bf16* Pw = Ps + wave * (16 * PSTR);

    for (int t0 = 0; t0 < T_SEQ; t0 += KVTILE) {
        __syncthreads();   // prior iteration's fragment reads done before restaging
        // stage Kc row-major, V transposed
        for (int i = 0; i < (KVTILE * D_MODEL) / 256; ++i) {
            int idx = i * 256 + tid;
            int r = idx >> 7, c = idx & 127;
            size_t g = ((size_t)(b * T_SEQ + t0 + r) * N_HEADS + n) * D_MODEL + c;
            Ks[r * KSTR + c]  = (bf16)(K[g] - meanK[head * D_MODEL + c]);
            Vt[c * VTSTR + r] = (bf16)V[g];
        }
        if (t0 + KVTILE < T_SEQ) {   // prefetch next tile (global_prefetch_b8)
            size_t gn = ((size_t)(b * T_SEQ + t0 + KVTILE + (tid >> 3)) * N_HEADS + n) * D_MODEL
                        + (size_t)(tid & 7) * 16;
            __builtin_prefetch(&K[gn], 0, 0);
            __builtin_prefetch(&V[gn], 0, 0);
        }
        __syncthreads();

        // S = Qc * Kc^T : 16 x 32 tile as two 16x16 accumulators
        v8f s0 = vzero, s1 = vzero;
#pragma unroll
        for (int kk = 0; kk < 4; ++kk) {
            Frag16 kb;
            load_b_frag(kb, Ks, KSTR, 0, kk * 32, lane);
            s0 = __builtin_amdgcn_wmma_f32_16x16x32_bf16(false, qa[kk].v, false, kb.v,
                                                         (short)0, s0, false, false);
            load_b_frag(kb, Ks, KSTR, 16, kk * 32, lane);
            s1 = __builtin_amdgcn_wmma_f32_16x16x32_bf16(false, qa[kk].v, false, kb.v,
                                                         (short)0, s1, false, false);
        }

        // online softmax: C layout -> row = j + 8*(lane>=16), col = lane&15
#pragma unroll
        for (int j = 0; j < 8; ++j) {
            float m = row16_max(fmaxf(s0[j], s1[j]));      // DPP butterfly, no LDS
            float newM = fmaxf(Mrow[j], m);
            float nm2  = newM * LOG2E;
            float corr = exp2f(Mrow[j] * LOG2E - nm2);
            Mrow[j] = newM;
            float p0 = exp2f(s0[j] * LOG2E - nm2);
            float p1 = exp2f(s1[j] * LOG2E - nm2);
            float rs = row16_sum(p0 + p1);                 // DPP butterfly, no LDS
            Lrow[j] = Lrow[j] * corr + rs;
#pragma unroll
            for (int c = 0; c < 8; ++c) O[c][j] *= corr;
            // spill P (bf16) to per-wave LDS scratch in row-major A-tile order
            int row = j + ((lane >> 4) << 3);
            Pw[row * PSTR + (lane & 15)]      = (bf16)p0;
            Pw[row * PSTR + 16 + (lane & 15)] = (bf16)p1;
        }
        // wave-local LDS ordering: stores above -> fragment loads below
        asm volatile("s_wait_dscnt 0" ::: "memory");

        Frag16 pa;
        load_a_frag(pa, Pw, PSTR, 0, 0, lane);
#pragma unroll
        for (int c = 0; c < 8; ++c) {
            Frag16 vb;
            load_b_frag(vb, Vt, VTSTR, c * 16, 0, lane);   // B(k,n) = V[kv=k][d=c*16+n]
            O[c] = __builtin_amdgcn_wmma_f32_16x16x32_bf16(false, pa.v, false, vb.v,
                                                           (short)0, O[c], false, false);
        }
    }

    // epilogue: normalize and store fp32
#pragma unroll
    for (int j = 0; j < 8; ++j) {
        int row = j + ((lane >> 4) << 3);
        float inv = 1.0f / Lrow[j];
        size_t g = ((size_t)(b * T_SEQ + q0 + wave * 16 + row) * N_HEADS + n) * D_MODEL
                   + (lane & 15);
#pragma unroll
        for (int c = 0; c < 8; ++c)
            out[g + c * 16] = O[c][j] * inv;
    }
}

extern "C" void kernel_launch(void* const* d_in, const int* in_sizes, int n_in,
                              void* d_out, int out_size, void* d_ws, size_t ws_size,
                              hipStream_t stream) {
    const float* Q    = (const float*)d_in[0];
    const float* K    = (const float*)d_in[1];
    const float* V    = (const float*)d_in[2];
    const float* stdv = (const float*)d_in[3];
    const float* tw   = (const float*)d_in[4];
    const float* tb   = (const float*)d_in[5];
    float* out = (float*)d_out;

    float* meanQ = (float*)d_ws;                                   // 64*128 f32
    float* meanK = meanQ + B_BATCH * N_HEADS * D_MODEL;            // 64*128 f32
    float* tau   = meanK + B_BATCH * N_HEADS * D_MODEL;            // 64 f32

    destat_stats_kernel<<<B_BATCH * N_HEADS, 256, 0, stream>>>(Q, K, stdv, tw, tb,
                                                               meanQ, meanK, tau);
    destat_attn_kernel<<<dim3(T_SEQ / QTILE, B_BATCH * N_HEADS), 256, 0, stream>>>(
        Q, K, V, meanQ, meanK, tau, out);
}
